// GPT_34016140984970
// MI455X (gfx1250) — matrix operand order repaired and checked by reading, not compiled, not run on hardware
//
#include <hip/hip_runtime.h>
#include <stdint.h>

#define TB 2
#define TS 1024
#define TV 50257
#define TD 768
#define TL 12
#define TH 12
#define THD 64
#define TF 3072

typedef __attribute__((ext_vector_type(16))) _Float16 v16h;
typedef __attribute__((ext_vector_type(8)))  _Float16 v8h;
typedef __attribute__((ext_vector_type(2)))  _Float16 v2h;
typedef __attribute__((ext_vector_type(8)))  float    v8f;
typedef __attribute__((ext_vector_type(4)))  uint32_t u32x4;
typedef __attribute__((ext_vector_type(8)))  int      i32x8;
typedef __attribute__((ext_vector_type(4)))  int      i32x4;

__device__ inline v8f wmma16(v16h a, v16h b, v8f c) {
  // D = A(16x32 f16) * B(32x16 f16) + C(16x16 f32)
  return __builtin_amdgcn_wmma_f32_16x16x32_f16(false, a, false, b, (short)0, c,
                                                false, false);
}

__device__ inline v16h concat8(v8h lo, v8h hi) {
  return __builtin_shufflevector(lo, hi, 0, 1, 2, 3, 4, 5, 6, 7,
                                 8, 9, 10, 11, 12, 13, 14, 15);
}

// Async 16B global -> LDS copy (GLOBAL_LOAD_ASYNC_TO_LDS_B128, ASYNCcnt)
__device__ inline void async_b128(const void* lds_ptr, const void* gptr) {
  uint32_t loff = (uint32_t)(uintptr_t)lds_ptr;   // LDS aperture: offset in addr[31:0]
  unsigned long long g = (unsigned long long)(uintptr_t)gptr;
  asm volatile("global_load_async_to_lds_b128 %0, %1, off"
               :: "v"(loff), "v"(g) : "memory");
}
__device__ inline void async_wait0() {
  asm volatile("s_wait_asynccnt 0" ::: "memory");
}
__device__ inline void tensor_wait0() {
#if __has_builtin(__builtin_amdgcn_s_wait_tensorcnt)
  __builtin_amdgcn_s_wait_tensorcnt(0);
#else
  asm volatile("s_wait_tensorcnt 0" ::: "memory");
#endif
}

// TDM: DMA a rows x 32-half tile (row stride lda halves) from global into LDS
// with 16B padding after each 64B row segment -> LDS stride 40 halves.
// D# built per CDNA5 ISA 8.3/8.4: data_size=2B, pad_interval=3 (16 dwords),
// pad_amount=3 (4 dwords), tile_dim0=32, tile_dim1=rows.
__device__ inline void tdm_load_tileA(const _Float16* Ag, int lda, int rows,
                                      _Float16* lds) {
#if __has_builtin(__builtin_amdgcn_tensor_load_to_lds)
  uint64_t ga = (uint64_t)(uintptr_t)Ag;
  u32x4 g0 = {};
  g0[0] = 1u;                                   // count=1, user descriptor
  g0[1] = (uint32_t)(uintptr_t)lds;             // lds_addr
  g0[2] = (uint32_t)ga;                         // global_addr[31:0]
  g0[3] = (uint32_t)(ga >> 32) | (2u << 30);    // global_addr[56:32] | type=2
  i32x8 g1 = {};
  g1[0] = (int)((1u << 16) | (1u << 20) | (3u << 22) | (3u << 25));
  //              data_size=2B  pad_en     pad_int=16dw  pad_amt=4dw
  g1[1] = (int)(((uint32_t)lda & 0xFFFFu) << 16);          // tensor_dim0[15:0]
  g1[2] = (int)(((uint32_t)lda >> 16) |
                (((uint32_t)rows & 0xFFFFu) << 16));       // dim0 hi | dim1 lo
  g1[3] = (int)(32u << 16);                                // dim1 hi=0 | tile_dim0=32
  g1[4] = rows;                                            // tile_dim1 (tile_dim2=0)
  g1[5] = lda;                                             // tensor_dim0_stride lo
  g1[6] = 0;
  g1[7] = 0;
  i32x4 z4 = {};
  i32x8 z8 = {};
  // 6-arg form (this toolchain): (g0, g1, g2, g3, g4, cpol)
  __builtin_amdgcn_tensor_load_to_lds(g0, g1, z4, z4, z8, 0);
#else
  // fallback: per-lane async copies (wave0 only calls this)
  int lane = threadIdx.x & 31;
  for (int ch = lane; ch < rows * 4; ch += 32) {
    int r = ch >> 2, c = (ch & 3) * 8;
    async_b128(&lds[r * 40 + c], &Ag[(size_t)r * lda + c]);
  }
  async_wait0();
#endif
}

// ---------------------------------------------------------------------------
// 128x64 GEMM tile core: C = A(f16, MxK row-major) * B(f32, KxN row-major)
// 256 threads = 8 waves (4 along M x 2 along N); wave tile 32x32 = 2x2 frags.
// A tile moved by TDM (one descriptor per K-step, issued by wave 0);
// B converted fp32->f16 and stored transposed (Bst[n][k], stride 40 halves)
// so every fragment load is a pair of contiguous ds_load_b128.
// ---------------------------------------------------------------------------
#define LDA_S 40
#define LDBT  40

__device__ inline void gemm128x64(const _Float16* __restrict__ Ag, int lda,
                                  const float* __restrict__ Bg, long ldb,
                                  int K, int nvalid,
                                  _Float16* As, _Float16* Bst,
                                  v8f acc[2][2]) {
  const int tid  = threadIdx.x;
  const int wave = tid >> 5;
  const int lane = tid & 31;
  const int wm = wave & 3;
  const int wn = wave >> 2;
  const int lr = lane & 15;
  const int hi = lane >> 4;

  for (int k0 = 0; k0 < K; k0 += 32) {
    // A tile 128x32 halves via Tensor Data Mover (wave 0 issues the DMA)
    if (wave == 0) tdm_load_tileA(Ag + k0, lda, 128, As);
    // B tile 32x64 fp32 -> f16, stored transposed: Bst[n*LDBT + k]
#pragma unroll
    for (int i = 0; i < 8; ++i) {
      int idx = tid + i * 256;       // 0..2047
      int r = idx >> 6;              // k row
      int c = idx & 63;              // n col
      float bv = (c < nvalid) ? Bg[(long)(k0 + r) * ldb + c] : 0.0f;
      Bst[c * LDBT + r] = (_Float16)bv;
    }
    if (k0 + 32 < K)
      __builtin_prefetch(&Bg[(long)(k0 + 32) * ldb], 0, 0);
    if (wave == 0) tensor_wait0();
    __syncthreads();

    const _Float16* Ar0 = &As[(wm * 32 + lr) * LDA_S];
    const _Float16* Ar1 = &As[(wm * 32 + 16 + lr) * LDA_S];
    // A-frag K map: e+16*hi -> 16 contiguous halves
    v16h a0 = concat8(*(const v8h*)(Ar0 + 16 * hi), *(const v8h*)(Ar0 + 16 * hi + 8));
    v16h a1 = concat8(*(const v8h*)(Ar1 + 16 * hi), *(const v8h*)(Ar1 + 16 * hi + 8));
    // B-frag K map: e+16*hi -> 16 contiguous halves in transposed layout
    const _Float16* Bc0 = &Bst[(wn * 32 + lr) * LDBT + 16 * hi];
    const _Float16* Bc1 = &Bst[(wn * 32 + 16 + lr) * LDBT + 16 * hi];
    v16h b0 = concat8(*(const v8h*)(Bc0), *(const v8h*)(Bc0 + 8));
    v16h b1 = concat8(*(const v8h*)(Bc1), *(const v8h*)(Bc1 + 8));

    acc[0][0] = wmma16(a0, b0, acc[0][0]);
    acc[0][1] = wmma16(a0, b1, acc[0][1]);
    acc[1][0] = wmma16(a1, b0, acc[1][0]);
    acc[1][1] = wmma16(a1, b1, acc[1][1]);
    __syncthreads();
  }
}

// ---------------------------------------------------------------------------
__global__ __launch_bounds__(256) void embed_kernel(const int* __restrict__ ids,
                                                    const float* __restrict__ embd,
                                                    const float* __restrict__ pos,
                                                    float* __restrict__ x) {
  int t  = blockIdx.x * 256 + threadIdx.x;
  int d  = t % TD;
  int bs = t / TD;
  int s  = bs % TS;
  x[t] = embd[(size_t)ids[bs] * TD + d] + pos[(size_t)s * TD + d];
}

// ---------------------------------------------------------------------------
__global__ __launch_bounds__(256) void ln_kernel(const float* __restrict__ x,
                                                 const float* __restrict__ g,
                                                 const float* __restrict__ bta,
                                                 _Float16* __restrict__ o) {
  __shared__ float red[256];
  const int row = blockIdx.x;
  const int tid = threadIdx.x;
  const float* xr = x + (size_t)row * TD;
  float v0 = xr[tid], v1 = xr[tid + 256], v2 = xr[tid + 512];
  red[tid] = v0 + v1 + v2;
  __syncthreads();
  for (int off = 128; off > 0; off >>= 1) {
    if (tid < off) red[tid] += red[tid + off];
    __syncthreads();
  }
  float mean = red[0] * (1.0f / TD);
  __syncthreads();
  float d0 = v0 - mean, d1 = v1 - mean, d2 = v2 - mean;
  red[tid] = d0 * d0 + d1 * d1 + d2 * d2;
  __syncthreads();
  for (int off = 128; off > 0; off >>= 1) {
    if (tid < off) red[tid] += red[tid + off];
    __syncthreads();
  }
  float rstd = rsqrtf(red[0] * (1.0f / TD) + 1e-5f);
  _Float16* orow = o + (size_t)row * TD;
  orow[tid]       = (_Float16)(d0 * rstd * g[tid]       + bta[tid]);
  orow[tid + 256] = (_Float16)(d1 * rstd * g[tid + 256] + bta[tid + 256]);
  orow[tid + 512] = (_Float16)(d2 * rstd * g[tid + 512] + bta[tid + 512]);
}

// ---------------------------------------------------------------------------
// QKV: grid (16, 3*H); one 128x64 tile per block (N=64 == one head)
// ---------------------------------------------------------------------------
__global__ __launch_bounds__(256) void qkv_kernel(const _Float16* __restrict__ xn,
                                                  const float* __restrict__ wq,
                                                  const float* __restrict__ wk,
                                                  const float* __restrict__ wv,
                                                  int l,
                                                  _Float16* __restrict__ q,
                                                  _Float16* __restrict__ k,
                                                  _Float16* __restrict__ v) {
  __shared__ _Float16 As[128 * LDA_S];
  __shared__ _Float16 Bst[64 * LDBT];
  const int row0 = blockIdx.x * 128;
  const int mat  = blockIdx.y / TH;
  const int hh   = blockIdx.y % TH;
  const float* W = (mat == 0 ? wq : (mat == 1 ? wk : wv)) +
                   ((size_t)l * TH + hh) * TD * THD;
  _Float16* Out = (mat == 0 ? q : (mat == 1 ? k : v));

  v8f zero = {};
  v8f acc[2][2];
#pragma unroll
  for (int i = 0; i < 2; ++i)
#pragma unroll
    for (int j = 0; j < 2; ++j) acc[i][j] = zero;

  gemm128x64(xn + (size_t)row0 * TD, TD, W, THD, TD, 64, As, Bst, acc);

  const int tid = threadIdx.x, wave = tid >> 5, lane = tid & 31;
  const int wm = wave & 3, wn = wave >> 2, lr = lane & 15, hi = lane >> 4;
#pragma unroll
  for (int i = 0; i < 2; ++i)
#pragma unroll
    for (int j = 0; j < 2; ++j)
#pragma unroll
      for (int r = 0; r < 8; ++r) {
        int row = row0 + wm * 32 + i * 16 + r + 8 * hi;
        int bb = row >> 10, s = row & (TS - 1);
        int col = wn * 32 + j * 16 + lr;
        Out[(((size_t)(bb * TH + hh)) * TS + s) * THD + col] =
            (_Float16)acc[i][j][r];
      }
}

// ---------------------------------------------------------------------------
// Flash attention: grid (S/128, H, B). 8 waves, each owns 16 query rows.
// K tile async-to-LDS [key][dim]; V transposed to [dim][key] during copy so
// both QK^T and PV B-fragments load as contiguous b128 pairs.
// ---------------------------------------------------------------------------
__global__ __launch_bounds__(256) void attn_kernel(const _Float16* __restrict__ q,
                                                   const _Float16* __restrict__ k,
                                                   const _Float16* __restrict__ v,
                                                   float* __restrict__ x) {
  __shared__ _Float16 Ks[32][72];     // [key][dim], 144 B rows (16B-aligned)
  __shared__ _Float16 Vst[64][LDBT];  // [dim][key]
  __shared__ _Float16 Ps[8][16 * 40]; // per-wave P staging, stride 40
  const int h = blockIdx.y, b = blockIdx.z;
  const int q0 = blockIdx.x * 128;
  const int tid = threadIdx.x, wave = tid >> 5, lane = tid & 31;
  const int lr = lane & 15, hi = lane >> 4;
  const size_t bh = ((size_t)b * TH + h) * TS;
  const _Float16* qp = q + bh * THD;
  const _Float16* kp = k + bh * THD;
  const _Float16* vp = v + bh * THD;
  const int qrow0 = q0 + wave * 16;

  // Q fragments (A layout): two contiguous 16B runs per frag, from global
  v16h qa0, qa1;
  {
    const _Float16* Qr = qp + (size_t)(qrow0 + lr) * THD;
    qa0 = concat8(*(const v8h*)(Qr + 16 * hi), *(const v8h*)(Qr + 16 * hi + 8));
    qa1 = concat8(*(const v8h*)(Qr + 32 + 16 * hi), *(const v8h*)(Qr + 40 + 16 * hi));
  }

  v8f zero = {};
  v8f o[4];
#pragma unroll
  for (int t = 0; t < 4; ++t) o[t] = zero;
  float mrow[8], lrow[8];
#pragma unroll
  for (int r = 0; r < 8; ++r) { mrow[r] = -1e30f; lrow[r] = 0.0f; }

  const int kend = q0 + 128;
  for (int kt = 0; kt < kend; kt += 32) {
    // K tile: 32x64 halves = 256 x 16B chunks, 1 per thread, async
    {
      int r = tid >> 3;
      int c = (tid & 7) * 8;
      async_b128(&Ks[r][c], &kp[(size_t)(kt + r) * THD + c]);
    }
    // V tile transposed through registers: Vst[dim][key]
#pragma unroll
    for (int i = 0; i < 4; ++i) {
      int idx = tid + i * 256;         // 0..1023 dword pairs
      int r = idx >> 5;                // key
      int c = (idx & 31) * 2;          // dim pair
      v2h pr = *(const v2h*)&vp[(size_t)(kt + r) * THD + c];
      Vst[c][r]     = pr[0];
      Vst[c + 1][r] = pr[1];
    }
    async_wait0();
    __syncthreads();

    // scores: S(16x32) = Q(16x64) * K^T(64x32)
    v8f sfr[2];
#pragma unroll
    for (int j = 0; j < 2; ++j) {
      const _Float16* Kr = &Ks[j * 16 + lr][0];
      v16h kb0 = concat8(*(const v8h*)(Kr + 16 * hi),
                         *(const v8h*)(Kr + 16 * hi + 8));
      v16h kb1 = concat8(*(const v8h*)(Kr + 32 + 16 * hi),
                         *(const v8h*)(Kr + 40 + 16 * hi));
      v8f sc = zero;
      sc = wmma16(qa0, kb0, sc);
      sc = wmma16(qa1, kb1, sc);
      sfr[j] = sc;
    }

    // scale + causal mask + row max
    float rmax[8], rs[8], sf[8];
#pragma unroll
    for (int r = 0; r < 8; ++r) {
      const int qrow = qrow0 + r + 8 * hi;
      float m = -1e30f;
#pragma unroll
      for (int j = 0; j < 2; ++j) {
        int key = kt + j * 16 + lr;
        float sv = (key <= qrow) ? sfr[j][r] * 0.125f : -1e30f;
        sfr[j][r] = sv;
        m = fmaxf(m, sv);
      }
      rmax[r] = m;
    }
#pragma unroll
    for (int off = 1; off < 16; off <<= 1)
#pragma unroll
      for (int r = 0; r < 8; ++r)
        rmax[r] = fmaxf(rmax[r], __shfl_xor(rmax[r], off, 32));

#pragma unroll
    for (int r = 0; r < 8; ++r) {
      float mn = fmaxf(mrow[r], rmax[r]);
      sf[r] = __expf(mrow[r] - mn);
      mrow[r] = mn;
      rs[r] = 0.0f;
    }
#pragma unroll
    for (int j = 0; j < 2; ++j)
#pragma unroll
      for (int r = 0; r < 8; ++r) {
        float pv = __expf(sfr[j][r] - mrow[r]);
        sfr[j][r] = pv;
        rs[r] += pv;
      }
#pragma unroll
    for (int off = 1; off < 16; off <<= 1)
#pragma unroll
      for (int r = 0; r < 8; ++r) rs[r] += __shfl_xor(rs[r], off, 32);
#pragma unroll
    for (int r = 0; r < 8; ++r) lrow[r] = lrow[r] * sf[r] + rs[r];
#pragma unroll
    for (int t = 0; t < 4; ++t)
#pragma unroll
      for (int r = 0; r < 8; ++r) o[t][r] *= sf[r];

    // stage P: D-frag layout -> A-frag layout (per-wave LDS, stride 40)
    _Float16* Pw = &Ps[wave][0];
#pragma unroll
    for (int j = 0; j < 2; ++j)
#pragma unroll
      for (int r = 0; r < 8; ++r)
        Pw[(r + 8 * hi) * 40 + j * 16 + lr] = (_Float16)sfr[j][r];
    __syncthreads();
    const _Float16* Pr = &Pw[lr * 40];
    v16h pa = concat8(*(const v8h*)(Pr + 16 * hi), *(const v8h*)(Pr + 16 * hi + 8));

    // O(16x64) += P(16x32) * V(32x64)
#pragma unroll
    for (int t = 0; t < 4; ++t) {
      const _Float16* Vr = &Vst[t * 16 + lr][16 * hi];
      v16h vbf = concat8(*(const v8h*)(Vr), *(const v8h*)(Vr + 8));
      o[t] = wmma16(pa, vbf, o[t]);
    }
    __syncthreads();
  }

  float inv[8];
#pragma unroll
  for (int r = 0; r < 8; ++r) inv[r] = 1.0f / lrow[r];
#pragma unroll
  for (int t = 0; t < 4; ++t)
#pragma unroll
    for (int r = 0; r < 8; ++r) {
      int qrow = qrow0 + r + 8 * hi;
      size_t idx = ((size_t)(b * TS + qrow)) * TD + h * THD + t * 16 + lr;
      x[idx] += o[t][r] * inv[r];
    }
}

// ---------------------------------------------------------------------------
__global__ __launch_bounds__(256) void ffn1_kernel(const _Float16* __restrict__ xn,
                                                   const float* __restrict__ w1,
                                                   const float* __restrict__ b1,
                                                   int l, _Float16* __restrict__ hb) {
  __shared__ _Float16 As[128 * LDA_S];
  __shared__ _Float16 Bst[64 * LDBT];
  const int row0 = blockIdx.x * 128;
  const int n0 = blockIdx.y * 64;
  const float* W = w1 + (size_t)l * TD * TF + n0;
  const float* bias = b1 + (size_t)l * TF + n0;

  v8f zero = {};
  v8f acc[2][2];
#pragma unroll
  for (int i = 0; i < 2; ++i)
#pragma unroll
    for (int j = 0; j < 2; ++j) acc[i][j] = zero;

  gemm128x64(xn + (size_t)row0 * TD, TD, W, TF, TD, 64, As, Bst, acc);

  const int tid = threadIdx.x, wave = tid >> 5, lane = tid & 31;
  const int wm = wave & 3, wn = wave >> 2, lr = lane & 15, hi = lane >> 4;
#pragma unroll
  for (int i = 0; i < 2; ++i)
#pragma unroll
    for (int j = 0; j < 2; ++j)
#pragma unroll
      for (int r = 0; r < 8; ++r) {
        int row = row0 + wm * 32 + i * 16 + r + 8 * hi;
        int col = wn * 32 + j * 16 + lr;
        float val = fmaxf(acc[i][j][r] + bias[col], 0.0f);
        hb[(size_t)row * TF + n0 + col] = (_Float16)val;
      }
}

// ---------------------------------------------------------------------------
__global__ __launch_bounds__(256) void ffn2_kernel(const _Float16* __restrict__ hb,
                                                   const float* __restrict__ w2,
                                                   const float* __restrict__ b2,
                                                   int l, float* __restrict__ x) {
  __shared__ _Float16 As[128 * LDA_S];
  __shared__ _Float16 Bst[64 * LDBT];
  const int row0 = blockIdx.x * 128;
  const int n0 = blockIdx.y * 64;
  const float* W = w2 + (size_t)l * TF * TD + n0;
  const float* bias = b2 + (size_t)l * TD + n0;

  v8f zero = {};
  v8f acc[2][2];
#pragma unroll
  for (int i = 0; i < 2; ++i)
#pragma unroll
    for (int j = 0; j < 2; ++j) acc[i][j] = zero;

  gemm128x64(hb + (size_t)row0 * TF, TF, W, TD, TF, 64, As, Bst, acc);

  const int tid = threadIdx.x, wave = tid >> 5, lane = tid & 31;
  const int wm = wave & 3, wn = wave >> 2, lr = lane & 15, hi = lane >> 4;
#pragma unroll
  for (int i = 0; i < 2; ++i)
#pragma unroll
    for (int j = 0; j < 2; ++j)
#pragma unroll
      for (int r = 0; r < 8; ++r) {
        int row = row0 + wm * 32 + i * 16 + r + 8 * hi;
        int col = wn * 32 + j * 16 + lr;
        x[(size_t)row * TD + n0 + col] += acc[i][j][r] + bias[col];
      }
}

// ---------------------------------------------------------------------------
__global__ __launch_bounds__(256) void logits_kernel(const _Float16* __restrict__ xf,
                                                     const float* __restrict__ outw,
                                                     const float* __restrict__ outb,
                                                     float* __restrict__ out) {
  __shared__ _Float16 As[128 * LDA_S];
  __shared__ _Float16 Bst[64 * LDBT];
  const int row0 = blockIdx.x * 128;
  const int n0 = blockIdx.y * 64;
  const int nvalid = (TV - n0 < 64) ? (TV - n0) : 64;

  v8f zero = {};
  v8f acc[2][2];
#pragma unroll
  for (int i = 0; i < 2; ++i)
#pragma unroll
    for (int j = 0; j < 2; ++j) acc[i][j] = zero;

  gemm128x64(xf + (size_t)row0 * TD, TD, outw + n0, TV, TD, nvalid, As, Bst, acc);

  const int tid = threadIdx.x, wave = tid >> 5, lane = tid & 31;
  const int wm = wave & 3, wn = wave >> 2, lr = lane & 15, hi = lane >> 4;
#pragma unroll
  for (int i = 0; i < 2; ++i)
#pragma unroll
    for (int j = 0; j < 2; ++j)
#pragma unroll
      for (int r = 0; r < 8; ++r) {
        int row = row0 + wm * 32 + i * 16 + r + 8 * hi;
        int col = wn * 32 + j * 16 + lr;
        if (col < nvalid)
          out[(size_t)row * TV + n0 + col] = acc[i][j][r] + outb[n0 + col];
      }
}

// ---------------------------------------------------------------------------
extern "C" void kernel_launch(void* const* d_in, const int* in_sizes, int n_in,
                              void* d_out, int out_size, void* d_ws, size_t ws_size,
                              hipStream_t stream) {
  const int*   ids  = (const int*)d_in[0];
  const float* embd = (const float*)d_in[1];
  const float* pos  = (const float*)d_in[2];
  const float* ln1g = (const float*)d_in[3];
  const float* ln1b = (const float*)d_in[4];
  const float* wq   = (const float*)d_in[5];
  const float* wk   = (const float*)d_in[6];
  const float* wv   = (const float*)d_in[7];
  const float* ln2g = (const float*)d_in[8];
  const float* ln2b = (const float*)d_in[9];
  const float* w1   = (const float*)d_in[10];
  const float* b1   = (const float*)d_in[11];
  const float* w2   = (const float*)d_in[12];
  const float* b2   = (const float*)d_in[13];
  const float* lnfg = (const float*)d_in[14];
  const float* lnfb = (const float*)d_in[15];
  const float* outw = (const float*)d_in[16];
  const float* outb = (const float*)d_in[17];
  float* out = (float*)d_out;

  char* p = (char*)d_ws;
  auto take = [&](size_t bytes) {
    char* r = p;
    p += (bytes + 255) & ~(size_t)255;
    return r;
  };
  float*    x  = (float*)take((size_t)TB * TS * TD * 4);
  _Float16* xn = (_Float16*)take((size_t)TB * TS * TD * 2);
  _Float16* qb = (_Float16*)take((size_t)TB * TS * TD * 2);
  _Float16* kb = (_Float16*)take((size_t)TB * TS * TD * 2);
  _Float16* vb = (_Float16*)take((size_t)TB * TS * TD * 2);
  _Float16* hb = (_Float16*)take((size_t)TB * TS * TF * 2);

  embed_kernel<<<dim3(TB * TS * TD / 256), 256, 0, stream>>>(ids, embd, pos, x);

  for (int l = 0; l < TL; ++l) {
    ln_kernel<<<dim3(TB * TS), 256, 0, stream>>>(x, ln1g + l * TD, ln1b + l * TD, xn);
    qkv_kernel<<<dim3(16, 3 * TH), 256, 0, stream>>>(xn, wq, wk, wv, l, qb, kb, vb);
    attn_kernel<<<dim3(TS / 128, TH, TB), 256, 0, stream>>>(qb, kb, vb, x);
    ln_kernel<<<dim3(TB * TS), 256, 0, stream>>>(x, ln2g + l * TD, ln2b + l * TD, xn);
    ffn1_kernel<<<dim3(16, TF / 64), 256, 0, stream>>>(xn, w1, b1, l, hb);
    ffn2_kernel<<<dim3(16, TD / 64), 256, 0, stream>>>(hb, w2, b2, l, x);
  }

  ln_kernel<<<dim3(TB * TS), 256, 0, stream>>>(x, lnfg, lnfb, xn);
  logits_kernel<<<dim3(16, (TV + 63) / 64), 256, 0, stream>>>(xn, outw, outb, out);
}
